// Cross_SelfAttention_85633057948147
// MI455X (gfx1250) — matrix operand order repaired and checked by reading, not compiled, not run on hardware
//
#include <hip/hip_runtime.h>
#include <hip/hip_bf16.h>

#define HW    4096
#define CCH   256
#define DQ    16

typedef __bf16 bf16_t;
typedef __attribute__((ext_vector_type(16))) __bf16 v16bf;
typedef __attribute__((ext_vector_type(8)))  float  v8f;
typedef __attribute__((ext_vector_type(4)))  unsigned int v4u;

__device__ inline bf16_t f2bf(float f) {
    union { float f; unsigned u; } in; in.f = f;
    unsigned r = (in.u + 0x7FFFu + ((in.u >> 16) & 1u)) >> 16;
    union { unsigned short s; bf16_t b; } out; out.s = (unsigned short)r;
    return out.b;
}

// ---------------------------------------------------------------------------
// 1x1 conv, 8 output channels per thread (x value re-used by 8 FMAs).
// out[d0+i, p] = bias[d0+i] + sum_c W[(d0+i)*256+c] * x[c*HW + p]
// grid = (HW/256, D/8), block = 256
// ---------------------------------------------------------------------------
__global__ void __launch_bounds__(256)
proj8_f32(const float* __restrict__ W, const float* __restrict__ bias,
          const float* __restrict__ xbase, float* __restrict__ out) {
    int p  = blockIdx.x * 256 + threadIdx.x;
    int d0 = blockIdx.y * 8;
    float acc[8];
    #pragma unroll
    for (int i = 0; i < 8; ++i) acc[i] = bias[d0 + i];
    for (int c = 0; c < CCH; ++c) {
        float xv = xbase[(size_t)c * HW + p];
        #pragma unroll
        for (int i = 0; i < 8; ++i)
            acc[i] = fmaf(W[(size_t)(d0 + i) * CCH + c], xv, acc[i]);
    }
    #pragma unroll
    for (int i = 0; i < 8; ++i) out[(size_t)(d0 + i) * HW + p] = acc[i];
}

__global__ void __launch_bounds__(256)
proj8_bf16(const float* __restrict__ W, const float* __restrict__ bias,
           const float* __restrict__ xbase, bf16_t* __restrict__ out) {
    int p  = blockIdx.x * 256 + threadIdx.x;
    int d0 = blockIdx.y * 8;
    float acc[8];
    #pragma unroll
    for (int i = 0; i < 8; ++i) acc[i] = bias[d0 + i];
    for (int c = 0; c < CCH; ++c) {
        float xv = xbase[(size_t)c * HW + p];
        #pragma unroll
        for (int i = 0; i < 8; ++i)
            acc[i] = fmaf(W[(size_t)(d0 + i) * CCH + c], xv, acc[i]);
    }
    #pragma unroll
    for (int i = 0; i < 8; ++i) out[(size_t)(d0 + i) * HW + p] = f2bf(acc[i]);
}

// ---------------------------------------------------------------------------
// Attention logits + row softmax -> P[i, j] (bf16)
// 8 rows per 1024-thread block: 128 threads per row, 32 logits per thread.
// k is read once per 8 rows (8x traffic cut vs 1 row/block).
// ---------------------------------------------------------------------------
#define TI 8
__global__ void __launch_bounds__(1024)
attn_softmax(const float* __restrict__ q, const float* __restrict__ k,
             bf16_t* __restrict__ P) {
    int i0   = blockIdx.x * TI;
    int tid  = threadIdx.x;          // 0..1023
    int r    = tid >> 7;             // row 0..7
    int tl   = tid & 127;            // lane within row-team
    int lane = tid & 31;
    int wv   = (tid >> 5) & 3;       // wave within row-team

    __shared__ float qs[TI][DQ];
    __shared__ float rbuf[TI][4];

    if (tid < TI * DQ)
        qs[tid >> 4][tid & 15] = q[(size_t)(tid & 15) * HW + i0 + (tid >> 4)];
    __syncthreads();

    float s[32];
    float m = -1e30f;
    #pragma unroll 4
    for (int ch = 0; ch < 32; ++ch) {
        int j = ch * 128 + tl;
        float acc = 0.f;
        #pragma unroll
        for (int d = 0; d < DQ; ++d)
            acc = fmaf(qs[r][d], k[(size_t)d * HW + j], acc);
        s[ch] = acc;
        m = fmaxf(m, acc);
    }
    for (int off = 16; off > 0; off >>= 1) m = fmaxf(m, __shfl_xor(m, off, 32));
    if (lane == 0) rbuf[r][wv] = m;
    __syncthreads();
    float bm = fmaxf(fmaxf(rbuf[r][0], rbuf[r][1]), fmaxf(rbuf[r][2], rbuf[r][3]));
    __syncthreads();

    float sum = 0.f;
    #pragma unroll
    for (int ch = 0; ch < 32; ++ch) {
        float e = __expf(s[ch] - bm);
        s[ch] = e;
        sum += e;
    }
    for (int off = 16; off > 0; off >>= 1) sum += __shfl_xor(sum, off, 32);
    if (lane == 0) rbuf[r][wv] = sum;
    __syncthreads();
    float inv = 1.f / (rbuf[r][0] + rbuf[r][1] + rbuf[r][2] + rbuf[r][3]);

    #pragma unroll 4
    for (int ch = 0; ch < 32; ++ch)
        P[(size_t)(i0 + r) * HW + ch * 128 + tl] = f2bf(s[ch] * inv);
}

// ---------------------------------------------------------------------------
// apply_attn: Out[c,i] = sum_j V[c,j] * P[i,j]  via v_wmma_f32_16x16x32_bf16
// Block = 8 waves = 2 i-tile groups x 4 channel-group waves.
// B operand (P tile, shared by the 4 cg waves) is double-buffered in LDS,
// filled with async global->LDS copies; A operand double-buffered in regs.
// ---------------------------------------------------------------------------
__global__ void __launch_bounds__(256)
apply_attn_wmma(const bf16_t* __restrict__ V, const bf16_t* __restrict__ P,
                float* __restrict__ Out) {
    __shared__ __align__(32) bf16_t pstage[2][2][16][64]; // [group][buf][row][k] = 8 KB

    int tid  = threadIdx.x;
    int lane = tid & 31;
    int wv   = tid >> 5;              // 0..7
    int g    = wv >> 2;               // i-tile group within block
    int cg   = wv & 3;                // channel group (64 channels)
    int i0   = (blockIdx.x * 2 + g) * 16;
    int c0   = cg * 64;
    int lm   = lane & 15, lh = lane >> 4;

    // async fill mapping: 128 lanes of the group cover 16 rows x 8 x 16B
    int fl   = cg * 32 + lane;        // 0..127 within group
    int frow = fl >> 3;
    int fkp  = fl & 7;
    const bf16_t* fillsrc = P + (size_t)(i0 + frow) * HW + fkp * 8;
    // low 32 bits of a flat shared-aperture address = LDS byte address (ISA 10.2)
    unsigned ldsdst0 = (unsigned)(uintptr_t)&pstage[g][0][frow][fkp * 8];
    unsigned ldsdst1 = (unsigned)(uintptr_t)&pstage[g][1][frow][fkp * 8];

    v8f acc[4] = {};
    const bf16_t* vrow0 = V + (size_t)(c0 + lm) * HW + 16 * lh;
    const bf16_t* brow  = &pstage[g][0][lm][16 * lh];   // buf stride = 1024 elems

    union AB { v4u u[2]; v16bf v; };
    AB a0[4], a1[4];

    auto loadA = [&](AB* a, int jk) {
        #pragma unroll
        for (int t = 0; t < 4; ++t) {
            const bf16_t* vr = vrow0 + (size_t)(t * 16) * HW + jk;
            a[t].u[0] = *(const v4u*)(vr);
            a[t].u[1] = *(const v4u*)(vr + 8);
        }
    };
    auto fill = [&](unsigned ldsdst, int j0) {
        asm volatile("global_load_async_to_lds_b128 %0, %1, off"
                     :: "v"(ldsdst), "v"(fillsrc + j0) : "memory");
    };

    fill(ldsdst0, 0);
    loadA(a0, 0);
    asm volatile("s_wait_asynccnt 0x0" ::: "memory");
    __syncthreads();

    for (int chunk = 0; chunk < 64; ++chunk) {
        int j0 = chunk * 64;
        int bb = chunk & 1;
        int jn = (j0 + 64) & (HW - 1);       // wraps on last iter (unused data)
        fill(bb ? ldsdst0 : ldsdst1, jn);    // refill other buffer

        const bf16_t* bp = brow + bb * 1024;

        // step 0: K = j0 .. j0+31
        loadA(a1, j0 + 32);
        v16bf b0 = *(const v16bf*)(bp);
        #pragma unroll
        for (int t = 0; t < 4; ++t)
            acc[t] = __builtin_amdgcn_wmma_f32_16x16x32_bf16(
                false, a0[t].v, false, b0, (short)0, acc[t], false, false);

        // step 1: K = j0+32 .. j0+63
        loadA(a0, jn);
        v16bf b1 = *(const v16bf*)(bp + 32);
        #pragma unroll
        for (int t = 0; t < 4; ++t)
            acc[t] = __builtin_amdgcn_wmma_f32_16x16x32_bf16(
                false, a1[t].v, false, b1, (short)0, acc[t], false, false);

        asm volatile("s_wait_asynccnt 0x0" ::: "memory");
        __syncthreads();
    }

    // D layout: acc[t][r] is Out[c0 + 16t + r + 8*lh, i0 + lm]
    #pragma unroll
    for (int t = 0; t < 4; ++t) {
        #pragma unroll
        for (int r = 0; r < 8; ++r) {
            int c = c0 + t * 16 + r + 8 * lh;
            Out[(size_t)c * HW + i0 + lm] = acc[t][r];
        }
    }
}

// ---------------------------------------------------------------------------
// final: y[d,p] = gamma*(sum_cc Wpt[d,cc]*outcat[cc,p] + bpt[d]) + x[d,p]
// 8 output channels per thread. grid = (HW/256, 256/8), block = 256
// ---------------------------------------------------------------------------
__global__ void __launch_bounds__(256)
final_proj8(const float* __restrict__ outcat, const float* __restrict__ Wpt,
            const float* __restrict__ bpt, const float* __restrict__ gamma,
            const float* __restrict__ x, float* __restrict__ y) {
    int p  = blockIdx.x * 256 + threadIdx.x;
    int d0 = blockIdx.y * 8;
    float acc[8];
    #pragma unroll
    for (int i = 0; i < 8; ++i) acc[i] = bpt[d0 + i];
    for (int cc = 0; cc < 512; ++cc) {
        float ov = outcat[(size_t)cc * HW + p];
        #pragma unroll
        for (int i = 0; i < 8; ++i)
            acc[i] = fmaf(Wpt[(size_t)(d0 + i) * 512 + cc], ov, acc[i]);
    }
    float gm = gamma[0];
    #pragma unroll
    for (int i = 0; i < 8; ++i) {
        size_t idx = (size_t)(d0 + i) * HW + p;
        y[idx] = gm * acc[i] + x[idx];
    }
}

extern "C" void kernel_launch(void* const* d_in, const int* in_sizes, int n_in,
                              void* d_out, int out_size, void* d_ws, size_t ws_size,
                              hipStream_t stream) {
    (void)in_sizes; (void)n_in; (void)out_size; (void)ws_size;
    const float* x     = (const float*)d_in[0];
    const float* Wq    = (const float*)d_in[1];
    const float* bq    = (const float*)d_in[2];
    const float* Wk    = (const float*)d_in[3];
    const float* bk    = (const float*)d_in[4];
    const float* Wv    = (const float*)d_in[5];
    const float* bv    = (const float*)d_in[6];
    const float* Wpt   = (const float*)d_in[7];
    const float* bpt   = (const float*)d_in[8];
    const float* gamma = (const float*)d_in[9];
    float* y = (float*)d_out;

    // workspace layout
    float*  qbuf   = (float*)d_ws;                          // 4 * 16*HW f32   (1 MB)
    float*  kbuf   = qbuf + (size_t)4 * DQ * HW;            // 4 * 16*HW f32   (1 MB)
    bf16_t* vbuf   = (bf16_t*)(kbuf + (size_t)4 * DQ * HW); // 4 * 256*HW bf16 (8 MB)
    float*  outcat = (float*)(vbuf + (size_t)4 * CCH * HW); // 512*HW f32      (8 MB)
    bf16_t* Pbuf   = (bf16_t*)(outcat + (size_t)512 * HW);  // HW*HW bf16      (32 MB)

    // ---- q, k, v projections for all 4 (batch, half) combos ----
    for (int b = 0; b < 2; ++b)
        for (int h = 0; h < 2; ++h) {
            int combo = b * 2 + h;
            const float* xb = x + ((size_t)b * 512 + (size_t)h * CCH) * HW;
            proj8_f32 <<<dim3(HW / 256, DQ / 8),  256, 0, stream>>>(Wq, bq, xb, qbuf + (size_t)combo * DQ * HW);
            proj8_f32 <<<dim3(HW / 256, DQ / 8),  256, 0, stream>>>(Wk, bk, xb, kbuf + (size_t)combo * DQ * HW);
            proj8_bf16<<<dim3(HW / 256, CCH / 8), 256, 0, stream>>>(Wv, bv, xb, vbuf + (size_t)combo * CCH * HW);
        }

    // ---- per (batch, half): softmax -> two WMMA GEMMs -> output projection ----
    for (int b = 0; b < 2; ++b)
        for (int h = 0; h < 2; ++h) {
            int combo = b * 2 + h;
            attn_softmax<<<HW / TI, 1024, 0, stream>>>(qbuf + (size_t)combo * DQ * HW,
                                                       kbuf + (size_t)combo * DQ * HW, Pbuf);
            // channels [0,256): V(b, half 0) @ P^T ; channels [256,512): V(b, half 1) @ P^T
            apply_attn_wmma<<<128, 256, 0, stream>>>(vbuf + (size_t)(b * 2 + 0) * CCH * HW, Pbuf, outcat);
            apply_attn_wmma<<<128, 256, 0, stream>>>(vbuf + (size_t)(b * 2 + 1) * CCH * HW, Pbuf,
                                                     outcat + (size_t)CCH * HW);
            const float* xo = x + ((size_t)b * 512 + (size_t)h * CCH) * HW;
            float*       yo = y + ((size_t)b * 512 + (size_t)h * CCH) * HW;
            final_proj8<<<dim3(HW / 256, CCH / 8), 256, 0, stream>>>(outcat, Wpt, bpt, gamma, xo, yo);
        }
}
